// GPTModel_22917945491933
// MI455X (gfx1250) — matrix operand order repaired and checked by reading, not compiled, not run on hardware
//
#include <hip/hip_runtime.h>

#define DEV __device__ __forceinline__

typedef __attribute__((ext_vector_type(16))) _Float16 v16h;
typedef __attribute__((ext_vector_type(8)))  _Float16 v8h;
typedef __attribute__((ext_vector_type(8)))  float    v8f;

namespace cfg {
constexpr int B = 2, T = 2048, C = 1024, H = 16, V = 32000, L = 3;
constexpr int HD = C / H;          // 64
constexpr int F  = 4 * C;          // 4096
constexpr int NT = B * T;          // 4096 tokens
constexpr float EPS   = 1e-5f;
constexpr float SCALE = 0.03125f;  // C^-0.5 (quirk: emb size, not head size)
}

// ------------------------------------------------------------- TDM (CDNA5 DMA)
#if __has_builtin(__builtin_amdgcn_tensor_load_to_lds) && \
    __has_builtin(__builtin_amdgcn_s_wait_tensorcnt)
#define ATHENA_HAS_TDM 1
typedef __attribute__((ext_vector_type(4))) unsigned int u32x4;
typedef __attribute__((ext_vector_type(8))) int i32x8;
typedef __attribute__((ext_vector_type(4))) int i32x4;

// 2D f16 tile load: tile_k contiguous halves per row, tile_rows rows with
// row stride `stride` (elements). LDS rows padded: pad_int_code/pad_amt_code
// per D# encoding (interval = 2^(v+1) DWORDs, amount = v+1 DWORDs).
DEV void tdm_load_tile_f16(const _Float16* lds_dst, const _Float16* gsrc,
                           int tile_k, int tile_rows, int stride,
                           int pad_int_code, int pad_amt_code) {
  unsigned long long ga = (unsigned long long)(const void*)gsrc;
  u32x4 g0;
  g0[0] = 1u;                                       // count=1, user descriptor
  g0[1] = (unsigned)(size_t)(const void*)lds_dst;   // low 32 bits == LDS offset
  g0[2] = (unsigned)ga;                             // global_addr[31:0]
  g0[3] = (unsigned)((ga >> 32) & 0x01ffffffull) | (2u << 30);  // addr[56:32]|type=2
  i32x8 g1;
  g1[0] = (1 << 16)                                 // data_size = 2 bytes
        | (1 << 20)                                 // pad_enable
        | (pad_int_code << 22) | (pad_amt_code << 25);
  g1[1] = 0;                                        // abar=0, tensor_dim0[15:0]=0
  g1[2] = 0x4000;                                   // tensor_dim0 = 1<<30
  g1[3] = 1 | (tile_k << 16);                       // tensor_dim1 = 1<<16, tile_dim0
  g1[4] = tile_rows;                                // tile_dim1 (tile_dim2 = 0)
  g1[5] = stride;                                   // tensor_dim0_stride[31:0]
  g1[6] = 0;
  g1[7] = 0;
  i32x4 z4 = {0, 0, 0, 0};
  i32x8 z8 = {0, 0, 0, 0, 0, 0, 0, 0};
  // 6-arg therock/clang-23 form: (g0, g1, g2, g3, extra, cpol)
  __builtin_amdgcn_tensor_load_to_lds(g0, g1, z4, z4, z8, 0);
}
DEV void tdm_wait() { __builtin_amdgcn_s_wait_tensorcnt(0); }
#else
#define ATHENA_HAS_TDM 0
DEV void tdm_load_tile_f16(const _Float16*, const _Float16*, int, int, int, int, int) {}
DEV void tdm_wait() {}
#endif

// ---------------------------------------------------------------- WMMA helpers
DEV v8f wmma16(v16h a, v16h b, v8f c) {
  return __builtin_amdgcn_wmma_f32_16x16x32_f16(false, a, false, b, (short)0, c,
                                                false, false);
}

// A-operand fragment (16x32 f16): lanes 0-15 -> M=lane, K 0-7 & 16-23;
// lanes 16-31 -> M=lane-16, K 8-15 & 24-31. LDS tile is [m][k] row-major.
DEV v16h ldfragA(const _Float16* base, int kp, int kk) {
  int lane = threadIdx.x & 31;
  const _Float16* p = base + (lane & 15) * kp + kk + ((lane >> 4) << 3);
  v8h lo = *(const v8h*)(p);
  v8h hi = *(const v8h*)(p + 16);
  v16h r;
#pragma unroll
  for (int i = 0; i < 8; ++i) { r[i] = lo[i]; r[i + 8] = hi[i]; }
  return r;
}

// B-operand fragment (32x16 f16): lanes 0-15 -> N=lane, K 0-7 & 16-23;
// lanes 16-31 -> N=lane-16, K 8-15 & 24-31. LDS tile is B^T, i.e. [n][k] row-major.
DEV v16h ldfragB(const _Float16* base, int kp, int kk) {
  int lane = threadIdx.x & 31;
  const _Float16* p = base + (lane & 15) * kp + kk + ((lane >> 4) << 3);
  v8h lo = *(const v8h*)(p);
  v8h hi = *(const v8h*)(p + 16);
  v16h r;
#pragma unroll
  for (int i = 0; i < 8; ++i) { r[i] = lo[i]; r[i + 8] = hi[i]; }
  return r;
}

DEV float xmax16(float v) {
#pragma unroll
  for (int m = 1; m <= 8; m <<= 1) v = fmaxf(v, __shfl_xor(v, m, 32));
  return v;
}
DEV float xsum16(float v) {
#pragma unroll
  for (int m = 1; m <= 8; m <<= 1) v += __shfl_xor(v, m, 32);
  return v;
}

// ---------------------------------------------------------------- small kernels
__global__ void embed_kernel(const int* __restrict__ x, const float* __restrict__ tok,
                             const float* __restrict__ pos, float* __restrict__ h) {
  using namespace cfg;
  size_t gid = (size_t)blockIdx.x * blockDim.x + threadIdx.x;
  int cc = (int)(gid % C);
  int bt = (int)(gid / C);           // b*T + t
  int t  = bt % T;
  h[gid] = tok[(size_t)x[bt] * C + cc] + pos[(size_t)t * C + cc];
}

// f16 weights stored TRANSPOSED: [l][j][c] with j = h*HD + d (N-major, K contig)
__global__ void repack_qkv_kernel(const float* __restrict__ Wq, const float* __restrict__ Wk,
                                  const float* __restrict__ Wv, _Float16* __restrict__ wq,
                                  _Float16* __restrict__ wk, _Float16* __restrict__ wv) {
  using namespace cfg;
  size_t gid = (size_t)blockIdx.x * blockDim.x + threadIdx.x;
  int d  = (int)(gid % HD);
  int c  = (int)((gid / HD) % C);
  int hh = (int)((gid / ((size_t)HD * C)) % H);
  int l  = (int)(gid / ((size_t)HD * C * H));
  size_t dst = ((size_t)l * C + (size_t)hh * HD + d) * C + c;
  wq[dst] = (_Float16)Wq[gid];
  wk[dst] = (_Float16)Wk[gid];
  wv[dst] = (_Float16)Wv[gid];
}

__global__ void f32_to_f16_kernel(const float* __restrict__ a, _Float16* __restrict__ o) {
  size_t gid = (size_t)blockIdx.x * blockDim.x + threadIdx.x;
  o[gid] = (_Float16)a[gid];
}

__global__ void zero_kernel(float* p) { p[0] = 0.f; }

// LayerNorm over the TIME axis (dim=1), unbiased variance; emits f16.
__global__ void ln_time_kernel(const float* __restrict__ h, _Float16* __restrict__ y16) {
  using namespace cfg;
  const int tx = threadIdx.x;                 // 0..31 : channel within group
  const int ty = threadIdx.y;                 // 0..7  : time slice
  const int b  = blockIdx.x / (C / 32);
  const int c  = (blockIdx.x % (C / 32)) * 32 + tx;
  const float* base = h + (size_t)b * T * C + c;
  float s = 0.f, s2 = 0.f;
  for (int t = ty; t < T; t += 8) {
    float v = base[(size_t)t * C];
    s += v; s2 += v * v;
  }
  __shared__ float rs[8][32], rq[8][32];
  rs[ty][tx] = s; rq[ty][tx] = s2;
  __syncthreads();
  if (ty == 0) {
    float a = 0.f, q = 0.f;
#pragma unroll
    for (int j = 0; j < 8; ++j) { a += rs[j][tx]; q += rq[j][tx]; }
    float mean = a / (float)T;
    float var  = (q - a * mean) / (float)(T - 1);   // unbiased (ddof=1)
    rs[0][tx] = mean;
    rq[0][tx] = rsqrtf(var + EPS);
  }
  __syncthreads();
  float mean = rs[0][tx], rstd = rq[0][tx];
  _Float16* yb = y16 + (size_t)b * T * C + c;
  for (int t = ty; t < T; t += 8)
    yb[(size_t)t * C] = (_Float16)((base[(size_t)t * C] - mean) * rstd);
}

// ---------------------------------------------------------------- tiled WMMA GEMM
// C[M,N] = A[M,K](f16) * B (+bias) (+resid) (+relu).
//   WF16=true : B is f16, PRE-TRANSPOSED [N][K] row-major (ldb = K); staged by
//               the Tensor Data Mover with double buffering when available.
//   WF16=false: B is f32 [K][N] row-major; converted inline while staging.
template <bool WF16, bool OUT16, bool RELU, bool BIAS, bool RESID>
__global__ __launch_bounds__(256) void gemm_kernel(
    const _Float16* __restrict__ A, const void* __restrict__ Bv,
    const float* __restrict__ bias, const float* resid, float* outf,
    _Float16* outh, int M, int N, int K, int lda, int ldb, int ldo) {
  constexpr int BM = 128, BN = 128, BK = 32, KP = BK + 8;
  constexpr bool TDM = WF16 && (ATHENA_HAS_TDM != 0);
  constexpr int NB = TDM ? 2 : 1;
  __shared__ alignas(16) _Float16 sA[NB * BM * KP];   // [m][k] (+pad)
  __shared__ alignas(16) _Float16 sB[NB * BN * KP];   // [n][k] (+pad)
  const int tid  = threadIdx.x;
  const int lane = tid & 31;
  const int wave = tid >> 5;
  const int bm = blockIdx.y * BM;
  const int bn = blockIdx.x * BN;
  const int wm = (wave >> 2) * 64;               // 2x4 wave grid, 64x32 per wave
  const int wn = (wave & 3) * 32;
  const _Float16* Bh = (const _Float16*)Bv;

  v8f acc[4][2] = {};
  int cur = 0;

  if constexpr (TDM) {
    if (wave == 0) {   // one wave posts DMA descriptors for the whole block
      tdm_load_tile_f16(sA, A + (size_t)bm * lda, BK, BM, lda, 3, 3);
      tdm_load_tile_f16(sB, Bh + (size_t)bn * ldb, BK, BN, ldb, 3, 3);
    }
  }

  for (int k0 = 0; k0 < K; k0 += BK) {
    if constexpr (TDM) {
      if (wave == 0) tdm_wait();
      __syncthreads();                     // current buffer ready for all waves
      if (wave == 0 && k0 + BK < K) {      // overlap next DMA with this compute
        int nb = cur ^ 1;
        tdm_load_tile_f16(sA + nb * BM * KP, A + (size_t)bm * lda + k0 + BK,
                          BK, BM, lda, 3, 3);
        tdm_load_tile_f16(sB + nb * BN * KP, Bh + (size_t)bn * ldb + k0 + BK,
                          BK, BN, ldb, 3, 3);
      }
    } else {
      // stage A: 128x32 halves (v8h per chunk)
#pragma unroll
      for (int it = 0; it < 2; ++it) {
        int c = tid + it * 256;
        int r = c >> 2, kk = (c & 3) * 8;
        *(v8h*)(sA + r * KP + kk) =
            *(const v8h*)(A + (size_t)(bm + r) * lda + k0 + kk);
      }
      if constexpr (WF16) {                // transposed f16 weights: same shape
#pragma unroll
        for (int it = 0; it < 2; ++it) {
          int c = tid + it * 256;
          int r = c >> 2, kk = (c & 3) * 8;
          *(v8h*)(sB + r * KP + kk) =
              *(const v8h*)(Bh + (size_t)(bn + r) * ldb + k0 + kk);
        }
      } else {                             // f32 [K][N]: convert + transpose
        const float* Bw = (const float*)Bv;
#pragma unroll
        for (int it = 0; it < 4; ++it) {
          int c = tid + it * 256;
          int kr = c >> 5, nc = (c & 31) * 4;
          float4 f = *(const float4*)(Bw + (size_t)(k0 + kr) * ldb + bn + nc);
          sB[(nc + 0) * KP + kr] = (_Float16)f.x;
          sB[(nc + 1) * KP + kr] = (_Float16)f.y;
          sB[(nc + 2) * KP + kr] = (_Float16)f.z;
          sB[(nc + 3) * KP + kr] = (_Float16)f.w;
        }
      }
      if (k0 + BK < K)   // speculative prefetch of next A tile into caches
        __builtin_prefetch(A + (size_t)(bm + (tid >> 1)) * lda + k0 + BK, 0, 1);
      __syncthreads();
    }

    const _Float16* cA = sA + cur * (BM * KP);
    const _Float16* cB = sB + cur * (BN * KP);
#pragma unroll
    for (int i = 0; i < 4; ++i) {
      v16h af = ldfragA(cA + (wm + i * 16) * KP, KP, 0);
#pragma unroll
      for (int j = 0; j < 2; ++j) {
        v16h bf = ldfragB(cB + (wn + j * 16) * KP, KP, 0);
        acc[i][j] = wmma16(af, bf, acc[i][j]);
      }
    }
    if constexpr (TDM) cur ^= 1;
    else __syncthreads();
  }

  // epilogue: C layout -> lanes 0-15 rows r, lanes 16-31 rows r+8
  const int nloc = lane & 15, mo = (lane >> 4) * 8;
#pragma unroll
  for (int i = 0; i < 4; ++i) {
#pragma unroll
    for (int j = 0; j < 2; ++j) {
      int gn = bn + wn + j * 16 + nloc;
      float bv = 0.f;
      if constexpr (BIAS) bv = bias[gn];
#pragma unroll
      for (int r = 0; r < 8; ++r) {
        int gm = bm + wm + i * 16 + mo + r;
        float v = acc[i][j][r] + bv;
        if constexpr (RESID) v += resid[(size_t)gm * ldo + gn];
        if constexpr (RELU)  v = fmaxf(v, 0.f);
        if constexpr (OUT16) outh[(size_t)gm * ldo + gn] = (_Float16)v;
        else                 outf[(size_t)gm * ldo + gn] = v;
      }
    }
  }
  (void)M; (void)N;
}

// ---------------------------------------------------------------- flash attention
// one block per (b, head, 128-row q tile); h += softmax(mask(QK^T*scale)) V
__global__ __launch_bounds__(256) void attn_kernel(
    const _Float16* __restrict__ q16, const _Float16* __restrict__ k16,
    const _Float16* __restrict__ vmat, float* __restrict__ h) {
  using namespace cfg;
  constexpr int KP = 72;
  __shared__ alignas(16) _Float16 sQ[128 * KP];       // [t][d]
  __shared__ alignas(16) _Float16 sK[64 * KP];        // [s][d]  == B^T for QK^T
  __shared__ alignas(16) _Float16 sVt[64 * KP];       // [d][s]  == B^T for PV
  __shared__ alignas(16) _Float16 sP[8 * 16 * KP];    // per-wave 16x64 P tile

  const int q0   = blockIdx.x * 128;
  const int head = blockIdx.y;
  const int b    = blockIdx.z;
  const int tid = threadIdx.x, lane = tid & 31, wave = tid >> 5;

  const _Float16* Q  = q16  + ((size_t)b * T) * C + head * HD;
  const _Float16* Kq = k16  + ((size_t)b * T) * C + head * HD;
  const _Float16* Vq = vmat + ((size_t)b * T) * C + head * HD;

#pragma unroll
  for (int it = 0; it < 4; ++it) {
    int c = tid + it * 256;
    int r = c >> 3, kk = (c & 7) * 8;
    *(v8h*)(sQ + r * KP + kk) = *(const v8h*)(Q + (size_t)(q0 + r) * C + kk);
  }

  float rm[8], rl[8];
#pragma unroll
  for (int r = 0; r < 8; ++r) { rm[r] = -3.0e38f; rl[r] = 0.f; }
  v8f o[4] = {};

  const int nloc = lane & 15, mo = (lane >> 4) * 8;
  const int nst = q0 / 64 + 2;          // s tiles up to and including diagonal
  for (int st = 0; st < nst; ++st) {
    const int s0 = st * 64;
    __syncthreads();
#pragma unroll
    for (int it = 0; it < 2; ++it) {
      int c = tid + it * 256;
      int r = c >> 3, kk = (c & 7) * 8;
      *(v8h*)(sK + r * KP + kk) = *(const v8h*)(Kq + (size_t)(s0 + r) * C + kk);
      v8h vv = *(const v8h*)(Vq + (size_t)(s0 + r) * C + kk);
#pragma unroll
      for (int j = 0; j < 8; ++j) sVt[(kk + j) * KP + r] = vv[j];
    }
    __syncthreads();

    // S = Q K^T (16 q-rows per wave, 64 s-cols)
    v8f sfr[4];
    {
      v16h a0 = ldfragA(sQ + (wave * 16) * KP, KP, 0);
      v16h a1 = ldfragA(sQ + (wave * 16) * KP, KP, 32);
#pragma unroll
      for (int j = 0; j < 4; ++j) {
        v8f z = {};
        z = wmma16(a0, ldfragB(sK + (j * 16) * KP, KP, 0), z);
        z = wmma16(a1, ldfragB(sK + (j * 16) * KP, KP, 32), z);
        sfr[j] = z;
      }
    }
    // scale + causal mask (in C layout)
#pragma unroll
    for (int j = 0; j < 4; ++j)
#pragma unroll
      for (int r = 0; r < 8; ++r) {
        int ss = s0 + j * 16 + nloc;
        int tt = q0 + wave * 16 + mo + r;
        sfr[j][r] = (ss > tt) ? -3.0e38f : sfr[j][r] * SCALE;
      }
    // online softmax: row stats live per half-wave (rows 0-7 / 8-15)
    float alpha[8], tsum[8];
#pragma unroll
    for (int r = 0; r < 8; ++r) {
      float m = fmaxf(fmaxf(sfr[0][r], sfr[1][r]), fmaxf(sfr[2][r], sfr[3][r]));
      m = xmax16(m);
      float nm = fmaxf(rm[r], m);
      alpha[r] = __expf(rm[r] - nm);
      rm[r] = nm;
      tsum[r] = 0.f;
    }
#pragma unroll
    for (int j = 0; j < 4; ++j)
#pragma unroll
      for (int r = 0; r < 8; ++r) {
        float p = __expf(sfr[j][r] - rm[r]);
        tsum[r] += p;
        sP[(wave * 16 + mo + r) * KP + j * 16 + nloc] = (_Float16)p;
      }
#pragma unroll
    for (int r = 0; r < 8; ++r) rl[r] = rl[r] * alpha[r] + xsum16(tsum[r]);
#pragma unroll
    for (int j = 0; j < 4; ++j)
#pragma unroll
      for (int r = 0; r < 8; ++r) o[j][r] *= alpha[r];
    __syncthreads();
    // O += P V
    {
      v16h p0 = ldfragA(sP + (wave * 16) * KP, KP, 0);
      v16h p1 = ldfragA(sP + (wave * 16) * KP, KP, 32);
#pragma unroll
      for (int j = 0; j < 4; ++j) {
        o[j] = wmma16(p0, ldfragB(sVt + (j * 16) * KP, KP, 0),  o[j]);
        o[j] = wmma16(p1, ldfragB(sVt + (j * 16) * KP, KP, 32), o[j]);
      }
    }
  }
  // h += O / l   (disjoint elements per block -> plain RMW)
  float* Ho = h + ((size_t)b * T) * C + head * HD;
#pragma unroll
  for (int j = 0; j < 4; ++j)
#pragma unroll
    for (int r = 0; r < 8; ++r) {
      int tt = q0 + wave * 16 + mo + r;
      int d  = j * 16 + nloc;
      float* p = Ho + (size_t)tt * C + d;
      *p += o[j][r] / rl[r];
    }
}

// ---------------------------------------------------------------- loss
__global__ __launch_bounds__(256) void loss_row_kernel(const float* __restrict__ logits,
                                                       const int* __restrict__ targets,
                                                       float* __restrict__ accum) {
  using namespace cfg;
  const int row = blockIdx.x;
  const float* p = logits + (size_t)row * V;
  const int tid = threadIdx.x;
  __shared__ float red[256];
  float m = -3.4e38f;
  for (int i = tid; i < V; i += 256) m = fmaxf(m, p[i]);
  red[tid] = m; __syncthreads();
  for (int s = 128; s > 0; s >>= 1) {
    if (tid < s) red[tid] = fmaxf(red[tid], red[tid + s]);
    __syncthreads();
  }
  m = red[0];
  __syncthreads();
  float sum = 0.f;
  for (int i = tid; i < V; i += 256) sum += __expf(p[i] - m);
  red[tid] = sum; __syncthreads();
  for (int s = 128; s > 0; s >>= 1) {
    if (tid < s) red[tid] += red[tid + s];
    __syncthreads();
  }
  if (tid == 0) {
    float nll = (m + __logf(red[0])) - p[targets[row]];
    atomicAdd(accum, nll);
  }
}

__global__ void loss_finish_kernel(const float* __restrict__ accum, float* __restrict__ out) {
  out[0] = accum[0] / (float)(cfg::B * cfg::T);
}

// ---------------------------------------------------------------- driver
extern "C" void kernel_launch(void* const* d_in, const int* in_sizes, int n_in,
                              void* d_out, int out_size, void* d_ws, size_t ws_size,
                              hipStream_t stream) {
  using namespace cfg;
  (void)in_sizes; (void)n_in; (void)out_size; (void)ws_size;
  const int*   x   = (const int*)  d_in[0];
  const int*   tgt = (const int*)  d_in[1];
  const float* tok = (const float*)d_in[2];
  const float* pos = (const float*)d_in[3];
  const float* Wq  = (const float*)d_in[4];
  const float* Wk  = (const float*)d_in[5];
  const float* Wv  = (const float*)d_in[6];
  const float* W1  = (const float*)d_in[7];
  const float* b1  = (const float*)d_in[8];
  const float* W2  = (const float*)d_in[9];
  const float* b2  = (const float*)d_in[10];
  const float* Wp  = (const float*)d_in[11];
  const float* bp  = (const float*)d_in[12];

  char* ws = (char*)d_ws;
  auto carve = [&](size_t bytes) {
    char* p = ws; ws += (bytes + 255) & ~(size_t)255; return p;
  };
  float*    h    = (float*)   carve((size_t)NT * C * 4);
  _Float16* y16  = (_Float16*)carve((size_t)NT * C * 2);     // also reused as h16
  char*     uni  = carve((size_t)NT * F * 2);                // union: qkv | ffn-mid
  _Float16* q16  = (_Float16*)uni;
  _Float16* k16  = q16 + (size_t)NT * C;
  _Float16* vb16 = k16 + (size_t)NT * C;
  _Float16* mid16= (_Float16*)uni;                           // alive after attention
  _Float16* wq16 = (_Float16*)carve((size_t)L * C * C * 2);  // transposed [l][j][c]
  _Float16* wk16 = (_Float16*)carve((size_t)L * C * C * 2);
  _Float16* wv16 = (_Float16*)carve((size_t)L * C * C * 2);
  float*    accum= (float*)   carve(256);
  _Float16* h16  = y16;

  float* logits = (float*)d_out;
  float* loss   = logits + (size_t)NT * V;

  embed_kernel<<<(int)(((size_t)NT * C) / 256), 256, 0, stream>>>(x, tok, pos, h);
  repack_qkv_kernel<<<(L * H * C * HD) / 256, 256, 0, stream>>>(Wq, Wk, Wv, wq16, wk16, wv16);
  zero_kernel<<<1, 1, 0, stream>>>(accum);

  dim3 lnb(32, 8);
  dim3 gq(C / 128, NT / 128);     // (8, 32)
  dim3 gf1(F / 128, NT / 128);    // (32, 32)
  dim3 gf2(C / 128, NT / 128);
  dim3 gat(T / 128, H, B);        // (16, 16, 2)

  for (int l = 0; l < L; ++l) {
    ln_time_kernel<<<B * (C / 32), lnb, 0, stream>>>(h, y16);
    gemm_kernel<true, true, false, false, false><<<gq, 256, 0, stream>>>(
        y16, wq16 + (size_t)l * C * C, nullptr, nullptr, nullptr, q16, NT, C, C, C, C, C);
    gemm_kernel<true, true, false, false, false><<<gq, 256, 0, stream>>>(
        y16, wk16 + (size_t)l * C * C, nullptr, nullptr, nullptr, k16, NT, C, C, C, C, C);
    gemm_kernel<true, true, false, false, false><<<gq, 256, 0, stream>>>(
        y16, wv16 + (size_t)l * C * C, nullptr, nullptr, nullptr, vb16, NT, C, C, C, C, C);
    attn_kernel<<<gat, 256, 0, stream>>>(q16, k16, vb16, h);
    ln_time_kernel<<<B * (C / 32), lnb, 0, stream>>>(h, y16);
    gemm_kernel<false, true, true, true, false><<<gf1, 256, 0, stream>>>(
        y16, W1 + (size_t)l * C * F, b1 + (size_t)l * F, nullptr, nullptr, mid16,
        NT, F, C, C, F, F);
    gemm_kernel<false, false, false, true, true><<<gf2, 256, 0, stream>>>(
        mid16, W2 + (size_t)l * F * C, b2 + (size_t)l * C, h, h, nullptr,
        NT, C, F, F, C, C);
  }

  f32_to_f16_kernel<<<(int)(((size_t)NT * C) / 256), 256, 0, stream>>>(h, h16);
  gemm_kernel<false, false, false, true, false><<<dim3(V / 128, NT / 128), 256, 0, stream>>>(
      h16, Wp, bp, nullptr, logits, nullptr, NT, V, C, C, V, V);
  loss_row_kernel<<<NT, 256, 0, stream>>>(logits, tgt, accum);
  loss_finish_kernel<<<1, 1, 0, stream>>>(accum, loss);
}
